// MetaMemoryGraphModel_50165218017406
// MI455X (gfx1250) — matrix operand order repaired and checked by reading, not compiled, not run on hardware
//
#include <hip/hip_runtime.h>
#include <hip/hip_bf16.h>
#include <math.h>

typedef __attribute__((ext_vector_type(16))) _Float16 v16h;
typedef __attribute__((ext_vector_type(8)))  float    v8f;

#define NUM_CATE 22
#define HID 512
#define MEM_ROWS 65536
#define TOPK 10

// ---------------------------------------------------------------------------
// Wave32 / block reduction helpers
// ---------------------------------------------------------------------------
__device__ __forceinline__ float wave_sum(float v) {
    for (int off = 16; off > 0; off >>= 1) v += __shfl_down(v, off, 32);
    return v;
}

__device__ __forceinline__ float block_sum(float v, float* red) {
    int lane = threadIdx.x & 31, w = threadIdx.x >> 5;
    v = wave_sum(v);
    if (lane == 0) red[w] = v;
    __syncthreads();
    int nw = blockDim.x >> 5;
    float s = (threadIdx.x < nw) ? red[threadIdx.x] : 0.0f;
    if (w == 0) s = wave_sum(s);
    if (threadIdx.x == 0) red[32] = s;
    __syncthreads();
    s = red[32];
    __syncthreads();
    return s;
}

// ---------------------------------------------------------------------------
// Fused embedding front-end (tiny: single workgroup)
// ---------------------------------------------------------------------------
__global__ void k_embed(const float* __restrict__ state,      // [512,49]
                        const float* __restrict__ tinfo,      // [22,5]
                        const float* __restrict__ tind,       // [22,1]
                        const float* __restrict__ tapp,       // [22,512]
                        const float* __restrict__ aprobs,     // [1,6]
                        const float* gdo1_w, const float* gdo1_b,
                        const float* gdo2_w, const float* gdo2_b,
                        const float* gdo3_w, const float* gdo3_b,
                        const float* gdo4_w, const float* gdo4_b,
                        const float* gdo5_w, const float* gdo5_b,
                        const float* gdf1_w, const float* gdf1_b,
                        const float* gdf2_w, const float* gdf2_b,
                        const float* ea_w,  const float* ea_b,
                        const float* conv1_w, const float* conv1_b,
                        const float* pw_w,  const float* pw_b,
                        float* __restrict__ img_out,          // d_out image region [64,49]
                        float* __restrict__ emb)              // ws [3136]
{
    __shared__ float sm[16000];                               // 64000 bytes LDS
    float* ti   = sm +     0;   // 132
    float* tA   = sm +   200;   // 484
    float* tB   = sm +   700;   // 484
    float* tgt  = sm +  1200;   // 22*518 = 11396
    float* h1   = sm + 12600;   // 22*128 = 2816
    float* te   = sm +     0;   // 22*49  = 1078 (reuse; ti/tA/tB dead)
    float* img  = sm +  1100;   // 3136   (tgt dead)
    float* xbuf = sm +  4300;   // 96*49 = 4704
    float* aemb = sm + 15500;   // 10
    const int tid = threadIdx.x, nt = blockDim.x;

    // ti = concat(target_info, target_indicator)  [22,6]
    for (int i = tid; i < NUM_CATE * 6; i += nt) {
        int r = i / 6, k = i % 6;
        ti[i] = (k < 5) ? tinfo[r * 5 + k] : tind[r];
    }
    __syncthreads();

    // t = relu(ti @ gdo1_w.T + b).T : t[r][c] = relu(dot(ti[c,:], w1[r,:]) + b1[r])
    for (int i = tid; i < NUM_CATE * NUM_CATE; i += nt) {
        int r = i / NUM_CATE, c = i % NUM_CATE;
        float acc = gdo1_b[r];
        for (int k = 0; k < 6; ++k) acc += ti[c * 6 + k] * gdo1_w[r * 6 + k];
        tA[i] = fmaxf(acc, 0.0f);
    }
    __syncthreads();

    const float* gw[4] = {gdo2_w, gdo3_w, gdo4_w, gdo5_w};
    const float* gb[4] = {gdo2_b, gdo3_b, gdo4_b, gdo5_b};
    float* pa = tA; float* pb = tB;
    for (int L = 0; L < 4; ++L) {
        for (int i = tid; i < NUM_CATE * NUM_CATE; i += nt) {
            int r = i / NUM_CATE, c = i % NUM_CATE;
            float acc = gb[L][c];
            for (int k = 0; k < NUM_CATE; ++k) acc += pa[r * NUM_CATE + k] * gw[L][c * NUM_CATE + k];
            pb[i] = fmaxf(acc, 0.0f);
        }
        __syncthreads();
        float* t = pa; pa = pb; pb = t;
    }
    // final t is in pa

    // tgt = concat((tapp.T @ t).T, tinfo, tind)  [22,518]
    for (int i = tid; i < NUM_CATE * 512; i += nt) {
        int c = i / 512, f = i % 512;
        float acc = 0.0f;
        for (int j = 0; j < NUM_CATE; ++j) acc += tapp[j * 512 + f] * pa[j * NUM_CATE + c];
        tgt[c * 518 + f] = acc;
    }
    for (int i = tid; i < NUM_CATE * 6; i += nt) {
        int c = i / 6, k = i % 6;
        tgt[c * 518 + 512 + k] = (k < 5) ? tinfo[c * 5 + k] : tind[c];
    }
    __syncthreads();

    // h1 = relu(tgt @ gdf1_w.T + b)  [22,128]
    for (int i = tid; i < NUM_CATE * 128; i += nt) {
        int c = i / 128, j = i % 128;
        float acc = gdf1_b[j];
        for (int k = 0; k < 518; ++k) acc += tgt[c * 518 + k] * gdf1_w[j * 518 + k];
        h1[i] = fmaxf(acc, 0.0f);
    }
    __syncthreads();

    // te = relu(h1 @ gdf2_w.T + b)  [22,49]
    for (int i = tid; i < NUM_CATE * 49; i += nt) {
        int c = i / 49, m = i % 49;
        float acc = gdf2_b[m];
        for (int k = 0; k < 128; ++k) acc += h1[c * 128 + k] * gdf2_w[m * 128 + k];
        te[i] = fmaxf(acc, 0.0f);
    }
    // action_emb = relu(aprobs @ ea_w.T + b)  [10]
    if (tid < 10) {
        float acc = ea_b[tid];
        for (int k = 0; k < 6; ++k) acc += aprobs[k] * ea_w[tid * 6 + k];
        aemb[tid] = fmaxf(acc, 0.0f);
    }
    __syncthreads();

    // image_embedding = relu(conv1x1(state))  [64,49]
    for (int i = tid; i < 64 * 49; i += nt) {
        int o = i / 49, hw = i % 49;
        float acc = conv1_b[o];
        for (int c = 0; c < 512; ++c) acc += conv1_w[o * 512 + c] * state[c * 49 + hw];
        float r = fmaxf(acc, 0.0f);
        img[i] = r;
        img_out[i] = r;
    }
    __syncthreads();

    // x = concat(img[64], te[22], broadcast(aemb)[10])  [96,49]
    for (int i = tid; i < 96 * 49; i += nt) {
        int c = i / 49, hw = i % 49;
        float v;
        if (c < 64)      v = img[c * 49 + hw];
        else if (c < 86) v = te[(c - 64) * 49 + hw];
        else             v = aemb[c - 86];
        xbuf[i] = v;
    }
    __syncthreads();

    // emb = relu(pw(x))  [64,49] -> ws flat [3136]
    for (int i = tid; i < 64 * 49; i += nt) {
        int o = i / 49, hw = i % 49;
        float acc = pw_b[o];
        for (int c = 0; c < 96; ++c) acc += pw_w[o * 96 + c] * xbuf[c * 49 + hw];
        emb[i] = fmaxf(acc, 0.0f);
    }
}

// ---------------------------------------------------------------------------
// LSTM gates: wave-per-output dual matvec
// ---------------------------------------------------------------------------
__global__ void k_gates(const float* __restrict__ x1, const float* __restrict__ W1, int K1,
                        const float* __restrict__ x2, const float* __restrict__ W2, int K2,
                        const float* __restrict__ b1, const float* __restrict__ b2,
                        float* __restrict__ out, int N)
{
    int lane = threadIdx.x & 31;
    int wave = (blockIdx.x * blockDim.x + threadIdx.x) >> 5;
    int nw   = (gridDim.x * blockDim.x) >> 5;
    for (int o = wave; o < N; o += nw) {
        const float* w1 = W1 + (size_t)o * K1;
        const float* w2 = W2 + (size_t)o * K2;
        float acc = 0.0f;
        for (int k = lane; k < K1; k += 32) acc += x1[k] * w1[k];
        for (int k = lane; k < K2; k += 32) acc += x2[k] * w2[k];
        acc = wave_sum(acc);
        if (lane == 0) out[o] = acc + b1[o] + b2[o];
    }
}

__device__ __forceinline__ float sigf(float x) { return 1.0f / (1.0f + expf(-x)); }

__global__ void k_lstm_combine(const float* __restrict__ gates, const float* __restrict__ cprev,
                               float* __restrict__ h_ws, float* __restrict__ h_out,
                               float* __restrict__ c_out)
{
    int i = blockIdx.x * blockDim.x + threadIdx.x;
    if (i < HID) {
        float ig = sigf(gates[i]);
        float fg = sigf(gates[HID + i]);
        float gg = tanhf(gates[2 * HID + i]);
        float og = sigf(gates[3 * HID + i]);
        float c = fg * cprev[i] + ig * gg;
        float h = og * tanhf(c);
        h_ws[i] = h; h_out[i] = h; c_out[i] = c;
    }
}

// ---------------------------------------------------------------------------
// Generic wave-per-output matvec: out[o] = act(dot(x, W[o,:]) + b[o])
// ---------------------------------------------------------------------------
__global__ void k_matvec(const float* __restrict__ x, const float* __restrict__ W,
                         const float* __restrict__ b, float* __restrict__ out,
                         int K, int N, int doRelu)
{
    int lane = threadIdx.x & 31;
    int wave = (blockIdx.x * blockDim.x + threadIdx.x) >> 5;
    int nw   = (gridDim.x * blockDim.x) >> 5;
    for (int o = wave; o < N; o += nw) {
        const float* wr = W + (size_t)o * K;
        float acc = 0.0f;
        for (int k = lane; k < K; k += 32) acc += x[k] * wr[k];
        acc = wave_sum(acc);
        if (lane == 0) {
            float r = acc + b[o];
            if (doRelu) r = fmaxf(r, 0.0f);
            out[o] = r;
        }
    }
}

// ---------------------------------------------------------------------------
// Single-workgroup LayerNorm (optionally fused ReLU); in-place safe
// ---------------------------------------------------------------------------
__global__ void k_ln(const float* __restrict__ in, float* __restrict__ out,
                     const float* __restrict__ g, const float* __restrict__ b,
                     int N, int doRelu)
{
    __shared__ float red[33];
    float s = 0.0f;
    for (int i = threadIdx.x; i < N; i += blockDim.x) s += in[i];
    float mean = block_sum(s, red) / (float)N;
    float v = 0.0f;
    for (int i = threadIdx.x; i < N; i += blockDim.x) { float d = in[i] - mean; v += d * d; }
    float var = block_sum(v, red) / (float)N;
    float rstd = rsqrtf(var + 1e-5f);
    for (int i = threadIdx.x; i < N; i += blockDim.x) {
        float r = (in[i] - mean) * rstd * g[i] + b[i];
        if (doRelu) r = fmaxf(r, 0.0f);
        out[i] = r;
    }
}

// ---------------------------------------------------------------------------
// Elementwise utilities
// ---------------------------------------------------------------------------
__global__ void k_addrelu(const float* a, const float* b, float* out, int n) {
    int i = blockIdx.x * blockDim.x + threadIdx.x;
    if (i < n) out[i] = fmaxf(a[i] + b[i], 0.0f);
}
__global__ void k_copy(const float* a, float* out, int n) {
    int i = blockIdx.x * blockDim.x + threadIdx.x;
    if (i < n) out[i] = a[i];
}
__global__ void k_zero(float* out, int n) {
    int i = blockIdx.x * blockDim.x + threadIdx.x;
    if (i < n) out[i] = 0.0f;
}

// ---------------------------------------------------------------------------
// One K=32 WMMA step: A = query chunk (from LDS, replicated across M rows),
// B = 16 memory rows (columns), accumulate into c.
// ---------------------------------------------------------------------------
__device__ __forceinline__ v8f wmma_step(const float* __restrict__ smq,
                                         const float* __restrict__ rowPtr,
                                         int k, int g, v8f c)
{
    // A layout (16-bit A 16x32): group g holds K = {g*8..g*8+7, 16+g*8..16+g*8+7}
    const float4* q4 = (const float4*)(smq + k + g * 8);
    float4 qa0 = q4[0];
    float4 qa1 = q4[1];
    float4 qb0 = q4[4];   // +16 floats
    float4 qb1 = q4[5];   // +20 floats
    // B layout (16-bit B 32x16): lane n holds K = {g*8..g*8+7, 16+g*8..16+g*8+7} of column n
    const float* rb = rowPtr + k + g * 8;
    float4 r0 = *(const float4*)(rb);
    float4 r1 = *(const float4*)(rb + 4);
    float4 r2 = *(const float4*)(rb + 16);
    float4 r3 = *(const float4*)(rb + 20);

    v16h a, bm;
    a[0]  = (_Float16)qa0.x; a[1]  = (_Float16)qa0.y; a[2]  = (_Float16)qa0.z; a[3]  = (_Float16)qa0.w;
    a[4]  = (_Float16)qa1.x; a[5]  = (_Float16)qa1.y; a[6]  = (_Float16)qa1.z; a[7]  = (_Float16)qa1.w;
    a[8]  = (_Float16)qb0.x; a[9]  = (_Float16)qb0.y; a[10] = (_Float16)qb0.z; a[11] = (_Float16)qb0.w;
    a[12] = (_Float16)qb1.x; a[13] = (_Float16)qb1.y; a[14] = (_Float16)qb1.z; a[15] = (_Float16)qb1.w;
    bm[0]  = (_Float16)r0.x; bm[1]  = (_Float16)r0.y; bm[2]  = (_Float16)r0.z; bm[3]  = (_Float16)r0.w;
    bm[4]  = (_Float16)r1.x; bm[5]  = (_Float16)r1.y; bm[6]  = (_Float16)r1.z; bm[7]  = (_Float16)r1.w;
    bm[8]  = (_Float16)r2.x; bm[9]  = (_Float16)r2.y; bm[10] = (_Float16)r2.z; bm[11] = (_Float16)r2.w;
    bm[12] = (_Float16)r3.x; bm[13] = (_Float16)r3.y; bm[14] = (_Float16)r3.z; bm[15] = (_Float16)r3.w;

    return __builtin_amdgcn_wmma_f32_16x16x32_f16(false, a, false, bm,
                                                  (short)0, c, false, false);
}

// ---------------------------------------------------------------------------
// Memory-attention scores via V_WMMA_F32_16X16X32_F16.
// Query q[512] staged into LDS with gfx1250 async-copy (ASYNCcnt-tracked),
// replicated across the 16 M-rows of A; each wave scores 16 memory rows.
// Two interleaved accumulators break the WMMA->WMMA RAW chain.
// ---------------------------------------------------------------------------
__global__ void k_scores_wmma(const float* __restrict__ q,
                              const float* __restrict__ rep,   // [65536,512]
                              float* __restrict__ scores)
{
    __shared__ float smq[HID];
    const int tid  = threadIdx.x;
    const int lane = tid & 31;
    const int wave = (blockIdx.x * blockDim.x + tid) >> 5;
    const int rowBase = wave * 16;
    const int n = lane & 15;     // column (memory row) within tile
    const int g = lane >> 4;     // lane half-group

    // Async-stage query into LDS: each of 256 lanes moves 8 bytes.
    {
        unsigned lds_addr = (unsigned)(uintptr_t)(&smq[tid * 2]);           // LDS byte offset
        unsigned long long gaddr = (unsigned long long)(uintptr_t)(q + tid * 2);
        asm volatile("global_load_async_to_lds_b64 %0, %1, off"
                     :: "v"(lds_addr), "v"(gaddr) : "memory");
        asm volatile("s_wait_asynccnt 0x0" ::: "memory");
    }
    __syncthreads();

    const float* rowPtr = rep + (size_t)(rowBase + n) * HID;
    v8f c0 = {}, c1 = {};
    for (int k = 0; k < HID; k += 64) {
        if (k + 192 < HID) __builtin_prefetch(rowPtr + k + 192, 0, 0);
        c0 = wmma_step(smq, rowPtr, k,      g, c0);
        c1 = wmma_step(smq, rowPtr, k + 32, g, c1);
    }
    v8f c = c0 + c1;
    if (g == 0) scores[rowBase + n] = c[0];   // D: VGPR0 lanes 0-15 => M=0, N=lane
}

// ---------------------------------------------------------------------------
// Top-10 + softmax + weighted gather (single workgroup; mutates scores)
// ---------------------------------------------------------------------------
__global__ void k_topk_gather(float* __restrict__ scores,
                              const float* __restrict__ sm,    // [65536,512]
                              const float* __restrict__ am,    // [65536,6]
                              float* __restrict__ att_state,   // [512]
                              float* __restrict__ att_acc,     // [518]
                              int M)
{
    __shared__ float svals[1024];
    __shared__ int   sidx[1024];
    __shared__ float topv[TOPK];
    __shared__ int   topi[TOPK];
    __shared__ float w[TOPK];
    const int tid = threadIdx.x;

    for (int t = 0; t < TOPK; ++t) {
        float best = -3.4e38f; int bi = 0;
        for (int i = tid; i < M; i += blockDim.x) {
            float v = scores[i];
            if (v > best) { best = v; bi = i; }
        }
        svals[tid] = best; sidx[tid] = bi;
        __syncthreads();
        for (int s = blockDim.x >> 1; s > 0; s >>= 1) {
            if (tid < s && svals[tid + s] > svals[tid]) {
                svals[tid] = svals[tid + s]; sidx[tid] = sidx[tid + s];
            }
            __syncthreads();
        }
        if (tid == 0) {
            topv[t] = svals[0]; topi[t] = sidx[0];
            scores[sidx[0]] = -3.4e38f;       // mask for next pass
        }
        __syncthreads();
    }
    if (tid == 0) {
        float mx = topv[0], s = 0.0f;
        for (int j = 0; j < TOPK; ++j) { w[j] = expf(topv[j] - mx); s += w[j]; }
        float inv = 1.0f / s;
        for (int j = 0; j < TOPK; ++j) w[j] *= inv;
    }
    __syncthreads();
    for (int d = tid; d < HID; d += blockDim.x) {
        float acc = 0.0f;
        for (int j = 0; j < TOPK; ++j) acc += w[j] * sm[(size_t)topi[j] * HID + d];
        att_state[d] = acc;
        att_acc[d]  += acc;
    }
    if (tid < 6) {
        float acc = 0.0f;
        for (int j = 0; j < TOPK; ++j) acc += w[j] * am[topi[j] * 6 + tid];
        att_acc[HID + tid] += acc;
    }
}

// ---------------------------------------------------------------------------
// Host launcher
// ---------------------------------------------------------------------------
extern "C" void kernel_launch(void* const* d_in, const int* in_sizes, int n_in,
                              void* d_out, int out_size, void* d_ws, size_t ws_size,
                              hipStream_t stream) {
    const float* in_state   = (const float*)d_in[0];
    const float* in_tinfo   = (const float*)d_in[1];
    const float* in_tind    = (const float*)d_in[2];
    const float* in_tapp    = (const float*)d_in[3];
    const float* in_aprobs  = (const float*)d_in[4];
    const float* in_hx      = (const float*)d_in[5];
    const float* in_cx      = (const float*)d_in[6];
    const float* in_rep     = (const float*)d_in[7];
    const float* in_smem    = (const float*)d_in[8];
    const float* in_amem    = (const float*)d_in[9];
#define P(i) ((const float*)d_in[10 + (i)])
    const float* conv1_w = P(0),  *conv1_b = P(1);
    const float* gdo1_w = P(2),  *gdo1_b = P(3);
    const float* gdo2_w = P(4),  *gdo2_b = P(5);
    const float* gdo3_w = P(6),  *gdo3_b = P(7);
    const float* gdo4_w = P(8),  *gdo4_b = P(9);
    const float* gdo5_w = P(10), *gdo5_b = P(11);
    const float* gdf1_w = P(12), *gdf1_b = P(13);
    const float* gdf2_w = P(14), *gdf2_b = P(15);
    const float* ea_w   = P(16), *ea_b   = P(17);
    const float* pw_w   = P(18), *pw_b   = P(19);
    const float* w_ih0  = P(20), *w_hh0  = P(21), *b_ih0 = P(22), *b_hh0 = P(23);
    const float* w_ih1  = P(24), *w_hh1  = P(25), *b_ih1 = P(26), *b_hh1 = P(27);
    const float* actor_w = P(28), *actor_b = P(29);
    const float* cr1_w = P(30), *cr1_b = P(31), *cr2_w = P(32), *cr2_b = P(33);
    const float* cse1_w = P(34), *cse1_b = P(35), *cse_ln1_g = P(36), *cse_ln1_b = P(37);
    const float* cse2_w = P(38), *cse2_b = P(39), *cse_ln2_g = P(40), *cse_ln2_b = P(41);
    const float* cae_w = P(42), *cae_b = P(43);
    const float* mme1_w = P(44), *mme1_b = P(45), *mme_ln1_g = P(46), *mme_ln1_b = P(47);
    const float* mme2_w = P(48), *mme2_b = P(49), *mme_ln2_g = P(50), *mme_ln2_b = P(51);
    const float* rb1_w = P(52), *rb1_b = P(53), *rb_ln1_g = P(54), *rb_ln1_b = P(55);
    const float* rb2_w = P(56), *rb2_b = P(57), *rb_ln2_g = P(58), *rb_ln2_b = P(59);
    const float* mp1_w = P(60), *mp1_b = P(61), *mp2_w = P(62), *mp2_b = P(63);
#undef P

    float* out = (float*)d_out;
    float* o_actor  = out + 0;
    float* o_critic = out + 6;
    float* o_hx     = out + 7;
    float* o_cx     = out + 1031;
    float* o_img    = out + 2055;
    float* o_csr    = out + 5191;
    float* o_lsm    = out + 5703;
    float* o_lam    = out + 6215;
    float* o_meta   = out + 6221;

    float* ws = (float*)d_ws;
    float* w_emb    = ws + 0;       // 3136
    float* w_gates  = ws + 3136;    // 2048 (reused by both layers)
    float* w_h0     = ws + 5184;    // 512
    float* w_xh     = ws + 5696;    // 512
    float* w_cse1   = ws + 6208;    // 256
    float* w_cse2   = ws + 6464;    // 512
    float* w_csr    = ws + 6976;    // 512
    float* w_att    = ws + 7488;    // 512
    float* w_acc    = ws + 8000;    // 518
    float* w_mme1   = ws + 8518;    // 1024
    float* w_mme2   = ws + 9542;    // 518
    float* w_meta   = ws + 10060;   // 1030
    float* w_rbt    = ws + 11090;   // 512
    float* w_res    = ws + 11602;   // 1030
    float* w_meta2  = ws + 12632;   // 1030
    float* w_mpt    = ws + 13662;   // 512
    float* w_crt    = ws + 14174;   // 64
    float* w_scores = ws + 14336;   // 65536

    // ---- embedding front-end (writes image_embedding + emb[3136]) ----
    k_embed<<<1, 256, 0, stream>>>(in_state, in_tinfo, in_tind, in_tapp, in_aprobs,
                                   gdo1_w, gdo1_b, gdo2_w, gdo2_b, gdo3_w, gdo3_b,
                                   gdo4_w, gdo4_b, gdo5_w, gdo5_b,
                                   gdf1_w, gdf1_b, gdf2_w, gdf2_b, ea_w, ea_b,
                                   conv1_w, conv1_b, pw_w, pw_b, o_img, w_emb);

    // ---- 2-layer LSTM ----
    k_gates<<<256, 256, 0, stream>>>(w_emb, w_ih0, 3136, in_hx, w_hh0, HID,
                                     b_ih0, b_hh0, w_gates, 4 * HID);
    k_lstm_combine<<<2, 256, 0, stream>>>(w_gates, in_cx, w_h0, o_hx, o_cx);
    k_gates<<<256, 256, 0, stream>>>(w_h0, w_ih1, HID, in_hx + HID, w_hh1, HID,
                                     b_ih1, b_hh1, w_gates, 4 * HID);
    k_lstm_combine<<<2, 256, 0, stream>>>(w_gates, in_cx + HID, w_xh, o_hx + HID, o_cx + HID);

    // ---- actor / critic / cae heads ----
    k_matvec<<<1, 256, 0, stream>>>(w_xh, actor_w, actor_b, o_actor, HID, 6, 0);
    k_matvec<<<8, 256, 0, stream>>>(w_xh, cr1_w, cr1_b, w_crt, HID, 64, 0);
    k_matvec<<<1, 256, 0, stream>>>(w_crt, cr2_w, cr2_b, o_critic, 64, 1, 0);
    k_matvec<<<1, 256, 0, stream>>>(in_aprobs, cae_w, cae_b, o_lam, 6, 6, 1);

    // ---- current_state_rep = relu(xh + cse(xh)) ----
    k_matvec<<<32, 256, 0, stream>>>(w_xh, cse1_w, cse1_b, w_cse1, HID, 256, 0);
    k_ln<<<1, 256, 0, stream>>>(w_cse1, w_cse1, cse_ln1_g, cse_ln1_b, 256, 1);
    k_matvec<<<64, 256, 0, stream>>>(w_cse1, cse2_w, cse2_b, w_cse2, 256, HID, 0);
    k_ln<<<1, 512, 0, stream>>>(w_cse2, w_cse2, cse_ln2_g, cse_ln2_b, HID, 0);
    k_addrelu<<<2, 256, 0, stream>>>(w_xh, w_cse2, w_csr, HID);
    k_copy<<<2, 256, 0, stream>>>(w_csr, o_csr, HID);
    k_copy<<<2, 256, 0, stream>>>(w_csr, o_lsm, HID);

    // ---- 4-head top-k memory attention (sequential heads; states_rep L2-resident) ----
    k_copy<<<2, 256, 0, stream>>>(w_csr, w_att, HID);
    k_zero<<<3, 256, 0, stream>>>(w_acc, HID + 6);
    for (int h = 0; h < 4; ++h) {
        // 65536 rows / 16 rows-per-wave / 8 waves-per-block = 512 blocks
        k_scores_wmma<<<512, 256, 0, stream>>>(w_att, in_rep, w_scores);
        k_topk_gather<<<1, 1024, 0, stream>>>(w_scores, in_smem, in_amem,
                                              w_att, w_acc, MEM_ROWS);
    }

    // ---- meta memory embedding ----
    k_matvec<<<128, 256, 0, stream>>>(w_acc, mme1_w, mme1_b, w_mme1, 518, 1024, 0);
    k_ln<<<1, 1024, 0, stream>>>(w_mme1, w_mme1, mme_ln1_g, mme_ln1_b, 1024, 1);
    k_matvec<<<65, 256, 0, stream>>>(w_mme1, mme2_w, mme2_b, w_mme2, 1024, 518, 0);
    k_ln<<<1, 512, 0, stream>>>(w_mme2, w_mme2, mme_ln2_g, mme_ln2_b, 518, 1);

    // ---- residual block + predict ----
    k_copy<<<2, 256, 0, stream>>>(w_csr, w_meta, HID);
    k_copy<<<3, 256, 0, stream>>>(w_mme2, w_meta + HID, 518);
    k_matvec<<<64, 256, 0, stream>>>(w_meta, rb1_w, rb1_b, w_rbt, 1030, HID, 0);
    k_ln<<<1, 512, 0, stream>>>(w_rbt, w_rbt, rb_ln1_g, rb_ln1_b, HID, 1);
    k_matvec<<<129, 256, 0, stream>>>(w_rbt, rb2_w, rb2_b, w_res, HID, 1030, 0);
    k_ln<<<1, 1024, 0, stream>>>(w_res, w_res, rb_ln2_g, rb_ln2_b, 1030, 0);
    k_addrelu<<<5, 256, 0, stream>>>(w_meta, w_res, w_meta2, 1030);
    k_matvec<<<64, 256, 0, stream>>>(w_meta2, mp1_w, mp1_b, w_mpt, 1030, HID, 1);
    k_matvec<<<1, 256, 0, stream>>>(w_mpt, mp2_w, mp2_b, o_meta, HID, 6, 0);

    (void)in_sizes; (void)n_in; (void)out_size; (void)ws_size;
}
